// HydraChannelMixer_90838558310605
// MI455X (gfx1250) — matrix operand order repaired and checked by reading, not compiled, not run on hardware
//
#include <hip/hip_runtime.h>
#include <hip/hip_bf16.h>
#include <math.h>

typedef __attribute__((ext_vector_type(16))) __bf16 v16bf;
typedef __attribute__((ext_vector_type(8)))  __bf16 v8bf;
typedef __attribute__((ext_vector_type(8)))  float  v8f;

// Problem constants (fixed by setup_inputs)
#define DD   256   // d_model
#define RR   32    // rank
#define HH   64    // hidden
#define BB   32
#define CC   64
#define PP   96
#define HS   260   // h tile LDS row stride (floats): 1040B rows, 16B-aligned, bank shift 4
#define HNS  272   // h_norm bf16 LDS row stride (544B rows, 16B aligned)
#define SS   33    // fp32 scratch row stride

// Assemble a 16-element bf16 A-fragment from its two contiguous 16B half-chunks.
__device__ __forceinline__ v16bf ld_afrag(const __bf16* p0) {
    v8bf lo = *(const v8bf*)p0;          // k = base .. base+7
    v8bf hi = *(const v8bf*)(p0 + 8);    // k = base+8 .. base+15
    return __builtin_shufflevector(lo, hi, 0,1,2,3,4,5,6,7,8,9,10,11,12,13,14,15);
}

__global__ __launch_bounds__(256)
void hydra_channel_mixer_kernel(
    const float* __restrict__ x,
    const float* __restrict__ ln_gamma, const float* __restrict__ ln_beta,
    const float* __restrict__ Wd, const float* __restrict__ bd,
    const float* __restrict__ Wq, const float* __restrict__ Wk,
    const float* __restrict__ Wv, const float* __restrict__ Wg,
    const float* __restrict__ bg,
    const float* __restrict__ Wu, const float* __restrict__ bu,
    const float* __restrict__ W1, const float* __restrict__ b1,
    const float* __restrict__ W2, const float* __restrict__ b2,
    float* __restrict__ out)
{
    // ---- LDS ----
    __shared__ __align__(16) float  lds_h[CC * HS];        // 66560 B pre-LN tile (residual)
    __shared__ __align__(32) float  lds_un[8704];          // 34816 B union: hn bf16 / QKVG fp32 scratch
    __shared__ __align__(32) __bf16 lds_wdf[DD * RR];      // 16384 B Wd, fragment-swizzled
    __shared__ __align__(32) __bf16 lds_wq4f[4 * RR * RR]; //  8192 B Wq,Wk,Wv,Wg fragment-swizzled
    __shared__ __align__(32) __bf16 lds_wuf[RR * DD];      // 16384 B Wu fragment-swizzled
    __shared__ __align__(32) __bf16 lds_hl[CC * RR];       //  4096 B h_low bf16
    __shared__ __align__(32) __bf16 lds_mi[CC * RR];       //  4096 B (h_attn*gate_c) bf16
    __shared__ float  lds_cv[DD];
    __shared__ float  lds_gate[DD];
    __shared__ float  lds_mu[CC], lds_rs[CC];
    __shared__ float  lds_qn[CC], lds_kn[CC];
    __shared__ float  lds_u[HH];
    __shared__ float  lds_gf[RR];

    const int t    = threadIdx.x;      // 0..255
    const int w    = t >> 5;           // wave 0..7
    const int lane = t & 31;
    const int ml   = lane & 15;
    const int lh   = lane >> 4;

    const int g = blockIdx.x;          // 0 .. B*P-1
    const int b = g / PP;
    const int p = g % PP;

    // -------- Phase A: async-copy x tile to LDS + fragment-swizzled weight staging --------
    {
        // Async global->LDS: thread t covers (row c4 = t>>6, 16B column chunk t&63).
        // 16 instructions/thread move the whole 64KB tile on ASYNCcnt, no VGPR writeback.
        const float* xrow0 = x + ((size_t)b * CC * PP + p) * DD;
        unsigned int loff0 = (unsigned int)(size_t)(lds_h) + (unsigned int)((t & 63) * 16);
        unsigned int goff0 = (unsigned int)((t & 63) * 16);
        #pragma unroll
        for (int i = 0; i < 16; ++i) {
            int c = i * 4 + (t >> 6);
            unsigned int loff = loff0 + (unsigned int)(c * (HS * 4));
            unsigned int goff = goff0 + (unsigned int)(c * (PP * DD * 4));
            asm volatile("global_load_async_to_lds_b128 %0, %1, %2"
                         :: "v"(loff), "v"(goff), "s"(xrow0) : "memory");
        }

        // Stage weights as bf16 in WMMA-B fragment layout:
        //   frag(kt,nt) stored as [lane][16e] contiguous; lane = lh*16 + (n&15),
        //   e = kb&15, lh = (kb>>4), kb = k&31  (contiguous k order, same as the ld_afrag A order)
        for (int i = 0; i < 32; ++i) {                 // Wd [256][32] -> 16 frags (8 kt x 2 nt)
            int idx = t + 256 * i;
            int n = idx & 31, k = idx >> 5;
            int kt = k >> 5, kb = k & 31;
            int dst = (((kt << 1) | (n >> 4)) * 32 + ((kb >> 4) << 4) + (n & 15)) * 16 + (kb & 15);
            lds_wdf[dst] = (__bf16)Wd[idx];
        }
        const float* Ws[4] = { Wq, Wk, Wv, Wg };
        for (int im = 0; im < 4; ++im)                 // W{q,k,v,g} [32][32] -> 2 frags each
            for (int i = 0; i < 4; ++i) {
                int idx = t + 256 * i;
                int n = idx & 31, k = idx >> 5;        // k = kb (K=32)
                int dst = (((im << 1) | (n >> 4)) * 32 + ((k >> 4) << 4) + (n & 15)) * 16 + (k & 15);
                lds_wq4f[dst] = (__bf16)Ws[im][idx];
            }
        for (int i = 0; i < 32; ++i) {                 // Wu [32][256] -> 16 frags (nt)
            int idx = t + 256 * i;
            int n = idx & 255, k = idx >> 8;           // k = kb (K=32)
            int dst = ((n >> 4) * 32 + ((k >> 4) << 4) + (n & 15)) * 16 + (k & 15);
            lds_wuf[dst] = (__bf16)Wu[idx];
        }
    }
    asm volatile("s_wait_asynccnt 0x0" ::: "memory");
    __syncthreads();

    // -------- Phase A2: column stats (chan_var, ddof=1) from LDS, conflict-free --------
    {
        float cs = 0.f, cs2 = 0.f;
        for (int c = 0; c < CC; ++c) {
            float v = lds_h[c * HS + t];
            cs += v; cs2 += v * v;
        }
        lds_cv[t] = (cs2 - cs * cs * (1.f / 64.f)) * (1.f / 63.f);
    }

    // -------- Phase B: per-row LN stats (wave-parallel shuffles) --------
    for (int rr = 0; rr < 8; ++rr) {
        int c = w * 8 + rr;
        float s = 0.f, s2 = 0.f;
        #pragma unroll
        for (int j = 0; j < 8; ++j) {
            float v = lds_h[c * HS + lane * 8 + j];
            s += v; s2 += v * v;
        }
        #pragma unroll
        for (int off = 16; off >= 1; off >>= 1) {
            s  += __shfl_xor(s,  off, 32);
            s2 += __shfl_xor(s2, off, 32);
        }
        if (lane == 0) {
            float mu  = s * (1.f / 256.f);
            float var = s2 * (1.f / 256.f) - mu * mu;
            lds_mu[c] = mu;
            lds_rs[c] = rsqrtf(var + 1e-5f);
        }
    }
    __syncthreads();

    // -------- Phase C: LayerNorm -> bf16 hn --------
    __bf16* hn = reinterpret_cast<__bf16*>(lds_un);
    {
        float gam = ln_gamma[t], bet = ln_beta[t];
        for (int c = 0; c < CC; ++c) {
            float v = (lds_h[c * HS + t] - lds_mu[c]) * lds_rs[c] * gam + bet;
            hn[c * HNS + t] = (__bf16)v;
        }
    }
    __syncthreads();

    // -------- Phase D: GEMM1  h_low = hn @ Wd + bd  ([64x256]@[256x32]) --------
    {
        int mt = w >> 1, nt = w & 1;                   // 4x2 tiles, one per wave
        const __bf16* arow = hn + (mt * 16 + ml) * HNS + lh * 16;
        v8f acc = {};
        #pragma unroll
        for (int kt = 0; kt < 8; ++kt) {
            v16bf af = ld_afrag(arow + kt * 32);
            v16bf bf = *(const v16bf*)(lds_wdf + (((kt << 1) | nt) * 32 + lane) * 16);
            acc = __builtin_amdgcn_wmma_f32_16x16x32_bf16(false, af, false, bf, (short)0, acc, false, false);
        }
        int n = nt * 16 + ml;
        float bdn = bd[n];
        #pragma unroll
        for (int r = 0; r < 8; ++r) {
            int m = mt * 16 + r + lh * 8;
            lds_hl[m * RR + n] = (__bf16)(acc[r] + bdn);
        }
    }
    __syncthreads();

    // -------- Phase E1: GEMM2  Q,K,V,G = h_low @ W{q,k,v,g}  (K=32, 1 WMMA each) --------
    float* Sq = lds_un;                                // hn is dead; reuse as fp32 scratch
    float* Sk = Sq + CC * SS;
    float* Sv = Sk + CC * SS;
    float* Sg = Sv + CC * SS;
    {
        int mt = w >> 1, nt = w & 1;
        int n = nt * 16 + ml;
        v16bf af = ld_afrag(lds_hl + (mt * 16 + ml) * RR + lh * 16);
        #pragma unroll
        for (int im = 0; im < 4; ++im) {
            v16bf bf = *(const v16bf*)(lds_wq4f + (((im << 1) | nt) * 32 + lane) * 16);
            v8f acc = {};
            acc = __builtin_amdgcn_wmma_f32_16x16x32_bf16(false, af, false, bf, (short)0, acc, false, false);
            float* S = Sq + im * (CC * SS);
            #pragma unroll
            for (int r = 0; r < 8; ++r)
                S[(mt * 16 + r + lh * 8) * SS + n] = acc[r];
        }
    }
    __syncthreads();

    // -------- Phase E2: l2norms, global feature, content gate -> mi (bf16) --------
    if (t < CC) {
        float qs = 0.f, ks = 0.f;
        for (int r2 = 0; r2 < RR; ++r2) {
            float q = Sq[t * SS + r2]; qs += q * q;
            float k = Sk[t * SS + r2]; ks += k * k;
        }
        lds_qn[t] = 1.f / fmaxf(sqrtf(qs), 1e-12f);
        lds_kn[t] = 1.f / fmaxf(sqrtf(ks), 1e-12f);
    }
    __syncthreads();
    if (t < RR) {
        float gf = 0.f;
        for (int c = 0; c < CC; ++c)
            gf += Sk[c * SS + t] * lds_kn[c] * Sv[c * SS + t];
        lds_gf[t] = gf;
    }
    __syncthreads();
    {
        float bgr = bg[t & 31];
        #pragma unroll
        for (int i = 0; i < 8; ++i) {
            int e = t + 256 * i;
            int c = e >> 5, r2 = e & 31;               // r2 == t&31 for all i
            float gc = 1.f / (1.f + __expf(-(Sg[c * SS + r2] + bgr)));
            float mi = Sq[c * SS + r2] * lds_qn[c] * lds_gf[r2] * gc;
            lds_mi[c * RR + r2] = (__bf16)mi;
        }
    }
    __syncthreads();

    // -------- Phase F: adaptive gate MLP (per-tile vector, VALU) --------
    if (t < HH) {
        float a = 0.f;
        for (int d2 = 0; d2 < DD; ++d2)
            a += lds_cv[d2] * W1[d2 * HH + t];
        a += b1[t];
        lds_u[t] = 0.5f * a * (1.f + erff(a * 0.70710678118654752f)); // exact gelu
    }
    __syncthreads();
    {
        float a = 0.f;
        for (int j = 0; j < HH; ++j)
            a += lds_u[j] * W2[j * DD + t];
        a += b2[t];
        lds_gate[t] = 1.f / (1.f + __expf(-a));
    }
    __syncthreads();

    // -------- Phase G: GEMM3  mixed = mi @ Wu ; fused +bu, xgate, +h residual, store --------
    for (int i = 0; i < 8; ++i) {
        int t2 = w * 8 + i;                            // 64 tiles: 4 (M) x 16 (N)
        int mt = t2 >> 4, nt = t2 & 15;
        v16bf af = ld_afrag(lds_mi + (mt * 16 + ml) * RR + lh * 16);
        v16bf bf = *(const v16bf*)(lds_wuf + (nt * 32 + lane) * 16);
        v8f acc = {};
        acc = __builtin_amdgcn_wmma_f32_16x16x32_bf16(false, af, false, bf, (short)0, acc, false, false);

        int d2 = nt * 16 + ml;
        float bun = bu[d2];
        float gt  = lds_gate[d2];
        #pragma unroll
        for (int r = 0; r < 8; ++r) {
            int c = mt * 16 + r + lh * 8;
            float hval = lds_h[c * HS + d2];
            size_t oi = ((size_t)(b * CC + c) * PP + p) * (size_t)DD + d2;
            out[oi] = hval + gt * (acc[r] + bun);
        }
    }
}

extern "C" void kernel_launch(void* const* d_in, const int* in_sizes, int n_in,
                              void* d_out, int out_size, void* d_ws, size_t ws_size,
                              hipStream_t stream) {
    (void)in_sizes; (void)n_in; (void)d_ws; (void)ws_size; (void)out_size;
    const float* x        = (const float*)d_in[0];
    const float* ln_gamma = (const float*)d_in[1];
    const float* ln_beta  = (const float*)d_in[2];
    const float* Wd       = (const float*)d_in[3];
    const float* bd       = (const float*)d_in[4];
    const float* Wq       = (const float*)d_in[5];
    const float* Wk       = (const float*)d_in[6];
    const float* Wv       = (const float*)d_in[7];
    const float* Wg       = (const float*)d_in[8];
    const float* bg       = (const float*)d_in[9];
    const float* Wu       = (const float*)d_in[10];
    const float* bu       = (const float*)d_in[11];
    const float* W1       = (const float*)d_in[12];
    const float* b1       = (const float*)d_in[13];
    const float* W2       = (const float*)d_in[14];
    const float* b2       = (const float*)d_in[15];
    float* out = (float*)d_out;

    dim3 grid(BB * PP);   // 3072 workgroups, one per (b,p) tile
    dim3 block(256);      // 8 waves (wave32)
    hydra_channel_mixer_kernel<<<grid, block, 0, stream>>>(
        x, ln_gamma, ln_beta, Wd, bd, Wq, Wk, Wv, Wg, bg, Wu, bu, W1, b1, W2, b2, out);
}
